// GNN_55293408969104
// MI455X (gfx1250) — matrix operand order repaired and checked by reading, compiled, not run on hardware
//
#include <hip/hip_runtime.h>

// ---------------------------------------------------------------------------
// GCN 2-layer + classifier for MI455X (gfx1250, wave32, WMMA)
// ---------------------------------------------------------------------------

#define N_NODES 100000
#define N_EDGES 1600000
#define KDIM    128      // F == H == 128 (all GEMM K dims)
#define HDIM    128
#define ODIM    40

typedef __attribute__((ext_vector_type(16))) __bf16 v16bf;
typedef __attribute__((ext_vector_type(8)))  float  v8f;

// ---------------------------------------------------------------------------
// GEMM: D[M x Ncols] = A[M x 128] * B[128 x Ncols] (+ bias), f32 in/out.
// bf16 hi/lo split -> 3x v_wmma_f32_16x16x32_bf16 per 32-wide K block.
// One wave per 16x16 tile; wave w of the block owns column tile w.
// M is a multiple of 16 (100000 = 6250*16). Column guards handle Ncols=40.
// ---------------------------------------------------------------------------
__global__ void __launch_bounds__(256)
gcn_gemm_wmma(const float* __restrict__ A, const float* __restrict__ B,
              const float* __restrict__ bias, float* __restrict__ D,
              int Ncols)
{
    const int lane  = threadIdx.x & 31;
    const int wave  = threadIdx.x >> 5;
    const int laneM = lane & 15;     // row (A) / col (B,D) within tile
    const int hi    = lane >> 4;     // half-wave selector

    const int m0 = blockIdx.x * 16;
    const int n0 = wave * 16;

    v8f acc = {};

    const int   ncol = n0 + laneM;
    const bool  nok  = (ncol < Ncols);
    const float* arow = A + (size_t)(m0 + laneM) * KDIM;

#pragma unroll
    for (int kb = 0; kb < KDIM; kb += 32) {
        v16bf ah, al, bh, bl;

        // --- A fragment: 16x32 bf16 layout (ISA 7.12.2) ------------------
        // lane<16: M=laneM, elems j<8 -> K=kb+j, j>=8 -> K=kb+16+(j-8)
        // lane>=16: same rows, K offset +8
#pragma unroll
        for (int j = 0; j < 16; ++j) {
            const int kk = ((j < 8) ? j : j + 8) + hi * 8;
            const float v  = arow[kb + kk];
            const __bf16 h = (__bf16)v;
            ah[j] = h;
            al[j] = (__bf16)(v - (float)h);
        }

        // --- B fragment: 32x16 bf16; N = laneM, K same order as A -------
#pragma unroll
        for (int j = 0; j < 16; ++j) {
            const int kk = kb + ((j < 8) ? j : j + 8) + hi * 8;
            const float v  = nok ? B[(size_t)kk * Ncols + ncol] : 0.0f;
            const __bf16 h = (__bf16)v;
            bh[j] = h;
            bl[j] = (__bf16)(v - (float)h);
        }

        // (Ah+Al)(Bh+Bl) ~= AhBh + AlBh + AhBl   (AlBl ~ 2^-16, dropped)
        acc = __builtin_amdgcn_wmma_f32_16x16x32_bf16(
                  false, ah, false, bh, (short)0, acc, false, false);
        acc = __builtin_amdgcn_wmma_f32_16x16x32_bf16(
                  false, al, false, bh, (short)0, acc, false, false);
        acc = __builtin_amdgcn_wmma_f32_16x16x32_bf16(
                  false, ah, false, bl, (short)0, acc, false, false);
    }

    if (nok) {
        const float bv = bias ? bias[ncol] : 0.0f;
#pragma unroll
        for (int r = 0; r < 8; ++r) {
            const int row = r + hi * 8;            // C/D layout: M = r + 8*hi
            D[(size_t)(m0 + row) * Ncols + ncol] = acc[r] + bv;
        }
    }
}

// ---------------------------------------------------------------------------
// Degree / normalization helpers
// ---------------------------------------------------------------------------
__global__ void zero_f32(float* __restrict__ p, int n)
{
    int i = blockIdx.x * blockDim.x + threadIdx.x;
    if (i < n) p[i] = 0.0f;
}

__global__ void deg_accum(const int* __restrict__ dst, float* __restrict__ deg)
{
    int e = blockIdx.x * blockDim.x + threadIdx.x;
    if (e < N_EDGES) atomicAdd(&deg[dst[e]], 1.0f);
}

__global__ void deg_to_dinv(float* __restrict__ deg, int n)
{
    int i = blockIdx.x * blockDim.x + threadIdx.x;
    if (i < n) deg[i] = rsqrtf(deg[i] + 1.0f);      // +1 self-loop
}

// ---------------------------------------------------------------------------
// agg[i][:] = h[i][:] * dinv[i]^2   (self-loop term; also initializes buffer)
// ---------------------------------------------------------------------------
__global__ void init_selfloop(const float* __restrict__ h,
                              const float* __restrict__ dinv,
                              float* __restrict__ out)
{
    unsigned i = blockIdx.x * blockDim.x + threadIdx.x;   // over N*128
    const float di = dinv[i >> 7];
    out[i] = h[i] * di * di;
}

// ---------------------------------------------------------------------------
// Edge scatter: one wave per edge, lane owns a float4 (32 lanes * 4 = 128).
// agg[dst] += h[src] * dinv[src]*dinv[dst]   via f32 atomics (L2-resident).
// ---------------------------------------------------------------------------
__global__ void __launch_bounds__(256)
edge_scatter(const float* __restrict__ h, const int* __restrict__ src,
             const int* __restrict__ dst, const float* __restrict__ dinv,
             float* __restrict__ out)
{
    const int lane = threadIdx.x & 31;
    const int e = blockIdx.x * (blockDim.x >> 5) + (threadIdx.x >> 5);
    if (e >= N_EDGES) return;

    const int s = src[e];
    const int d = dst[e];
    const float w = dinv[s] * dinv[d];

    const float4 v = ((const float4*)(h + (size_t)s * HDIM))[lane];
    float* o = (float*)(out + (size_t)d * HDIM) + lane * 4;
    atomicAdd(o + 0, v.x * w);
    atomicAdd(o + 1, v.y * w);
    atomicAdd(o + 2, v.z * w);
    atomicAdd(o + 3, v.w * w);
}

// ---------------------------------------------------------------------------
// In-place bias + ReLU over N*128 features
// ---------------------------------------------------------------------------
__global__ void bias_relu(float* __restrict__ g, const float* __restrict__ b)
{
    unsigned i = blockIdx.x * blockDim.x + threadIdx.x;
    const float v = g[i] + b[i & (HDIM - 1)];
    g[i] = v > 0.0f ? v : 0.0f;
}

// ---------------------------------------------------------------------------
static inline size_t align256(size_t x) { return (x + 255) & ~(size_t)255; }

extern "C" void kernel_launch(void* const* d_in, const int* in_sizes, int n_in,
                              void* d_out, int out_size, void* d_ws, size_t ws_size,
                              hipStream_t stream)
{
    (void)in_sizes; (void)n_in; (void)out_size; (void)ws_size;

    const float* x  = (const float*)d_in[0];
    const int*   ei = (const int*)  d_in[1];   // [2, E] row-major: src then dst
    const float* W1 = (const float*)d_in[2];
    const float* b1 = (const float*)d_in[3];
    const float* W2 = (const float*)d_in[4];
    const float* b2 = (const float*)d_in[5];
    const float* Wc = (const float*)d_in[6];
    const float* bc = (const float*)d_in[7];
    const int* src = ei;
    const int* dst = ei + N_EDGES;

    char* ws = (char*)d_ws;
    float* dinv = (float*)ws;                                  // N floats
    size_t off  = align256((size_t)N_NODES * sizeof(float));
    float* L    = (float*)(ws + off);                          // N*128 (linear out)
    off        += align256((size_t)N_NODES * HDIM * sizeof(float));
    float* G    = (float*)(ws + off);                          // N*128 (aggregate)

    const int NH = N_NODES * HDIM;                             // 12.8M
    const dim3 b256(256);
    const dim3 gNode((N_NODES + 255) / 256);
    const dim3 gEdge((N_EDGES + 255) / 256);
    const dim3 gEdgeWave((N_EDGES + 7) / 8);                   // 8 waves/block
    const dim3 gFeat(NH / 256);
    const dim3 gGemm(N_NODES / 16);                            // 6250 row tiles

    // normalization: deg -> dinv = rsqrt(deg+1)
    zero_f32   <<<gNode, b256, 0, stream>>>(dinv, N_NODES);
    deg_accum  <<<gEdge, b256, 0, stream>>>(dst, dinv);
    deg_to_dinv<<<gNode, b256, 0, stream>>>(dinv, N_NODES);

    // ---- layer 1: L = x @ W1 ; G = aggregate(L) ; G = relu(G + b1) ----
    gcn_gemm_wmma<<<gGemm, 256, 0, stream>>>(x, W1, nullptr, L, HDIM);
    init_selfloop<<<gFeat, b256, 0, stream>>>(L, dinv, G);
    edge_scatter <<<gEdgeWave, b256, 0, stream>>>(L, src, dst, dinv, G);
    bias_relu    <<<gFeat, b256, 0, stream>>>(G, b1);

    // ---- layer 2: L = G @ W2 ; G = aggregate(L) ; G = relu(G + b2) ----
    gcn_gemm_wmma<<<gGemm, 256, 0, stream>>>(G, W2, nullptr, L, HDIM);
    init_selfloop<<<gFeat, b256, 0, stream>>>(L, dinv, G);
    edge_scatter <<<gEdgeWave, b256, 0, stream>>>(L, src, dst, dinv, G);
    bias_relu    <<<gFeat, b256, 0, stream>>>(G, b2);

    // ---- head: out = G @ Wc + bc  (Ncols=40 -> 3 column tiles, 3 waves) ----
    gcn_gemm_wmma<<<gGemm, 96, 0, stream>>>(G, Wc, bc, (float*)d_out, ODIM);
}
